// LlamaSparseBlock_12266426597393
// MI455X (gfx1250) — matrix operand order repaired and checked by reading, not compiled, not run on hardware
//
#include <hip/hip_runtime.h>

// ---------------------------------------------------------------------------
// LlamaSparseBlock on MI455X (gfx1250): bf16 WMMA for all GEMMs + flash attn.
// Async global->LDS staging for conversion-free streams (A tiles, Q/K tiles).
// ---------------------------------------------------------------------------

#define T_SEQ 2048
#define HID   4096
#define NHEAD 32
#define HEADD 128
#define ISZ   11008
#define ATTNK 16
#define MLPK  2048

typedef __attribute__((ext_vector_type(16))) __bf16 v16bf;
typedef __attribute__((ext_vector_type(8)))  __bf16 v8bf;
typedef __attribute__((ext_vector_type(8)))  float  v8f;

static __device__ __forceinline__ v16bf mkv16(v8bf lo, v8bf hi) {
  v16bf r;
#pragma unroll
  for (int i = 0; i < 8; ++i) { r[i] = lo[i]; r[i + 8] = hi[i]; }
  return r;
}

static __device__ __forceinline__ v8f wmma_bf16(v16bf a, v16bf b, v8f c) {
  // D = A(16x32 bf16) * B(32x16 bf16) + C(16x16 f32)
  return __builtin_amdgcn_wmma_f32_16x16x32_bf16(false, a, false, b,
                                                 (short)0, c, false, false);
}

// --- gfx1250 async global->LDS copies (ASYNCcnt) ---------------------------
// INST_OFFSET is added to BOTH the global and LDS addresses (ISA 08 §4.4),
// so one LDS-address VGPR + one 64-bit vaddr covers a contiguous run.
static __device__ __forceinline__ unsigned lds_off32(const void* p) {
  // generic LDS pointer = {shared aperture, 32-bit LDS offset}; truncate.
  return (unsigned)(unsigned long long)p;
}
static __device__ __forceinline__ void async_ld_b128x2(unsigned lds, const void* g) {
  asm volatile(
      "global_load_async_to_lds_b128 %0, %1, off\n\t"
      "global_load_async_to_lds_b128 %0, %1, off offset:16"
      :: "v"(lds), "v"(g) : "memory");
}
static __device__ __forceinline__ void async_ld_b128x8(unsigned lds, const void* g) {
  asm volatile(
      "global_load_async_to_lds_b128 %0, %1, off\n\t"
      "global_load_async_to_lds_b128 %0, %1, off offset:16\n\t"
      "global_load_async_to_lds_b128 %0, %1, off offset:32\n\t"
      "global_load_async_to_lds_b128 %0, %1, off offset:48\n\t"
      "global_load_async_to_lds_b128 %0, %1, off offset:64\n\t"
      "global_load_async_to_lds_b128 %0, %1, off offset:80\n\t"
      "global_load_async_to_lds_b128 %0, %1, off offset:96\n\t"
      "global_load_async_to_lds_b128 %0, %1, off offset:112"
      :: "v"(lds), "v"(g) : "memory");
}
static __device__ __forceinline__ void wait_async0() {
  asm volatile("s_wait_asynccnt 0x0" ::: "memory");
}

// ---------------------------------------------------------------------------
// RMSNorm: fp32 in -> bf16 out (one block per row)
// ---------------------------------------------------------------------------
__global__ __launch_bounds__(256) void rmsnorm_kernel(
    const float* __restrict__ x, const float* __restrict__ w,
    __bf16* __restrict__ out, int H) {
  __shared__ float red[256];
  const int t = blockIdx.x;
  const float* xr = x + (size_t)t * H;
  float ss = 0.0f;
  for (int i = threadIdx.x; i < H; i += 256) { float v = xr[i]; ss += v * v; }
  red[threadIdx.x] = ss;
  __syncthreads();
  for (int s = 128; s > 0; s >>= 1) {
    if (threadIdx.x < s) red[threadIdx.x] += red[threadIdx.x + s];
    __syncthreads();
  }
  const float inv = rsqrtf(red[0] / (float)H + 1e-6f);
  __bf16* orow = out + (size_t)t * H;
  for (int i = threadIdx.x; i < H; i += 256)
    orow[i] = (__bf16)(xr[i] * inv * w[i]);
}

// ---------------------------------------------------------------------------
// Attention head predictor: 32 logits/row, top-16 mask via O(32^2) wave rank.
// ---------------------------------------------------------------------------
__global__ __launch_bounds__(32) void attn_pred_kernel(
    const __bf16* __restrict__ hs, const float* __restrict__ w,
    const float* __restrict__ b, float* __restrict__ head_mask) {
  const int t = blockIdx.x;
  const int o = threadIdx.x;
  const __bf16* hr = hs + (size_t)t * HID;
  const float* wr = w + (size_t)o * HID;
  float acc = 0.0f;
  for (int i = 0; i < HID; ++i) acc += (float)hr[i] * wr[i];
  acc += b[o];
  int rank = 0;
#pragma unroll
  for (int j = 0; j < 32; ++j) {
    float lj = __shfl(acc, j, 32);
    if (lj > acc || (lj == acc && j < o)) rank++;
  }
  head_mask[t * NHEAD + o] = (rank < ATTNK) ? 1.0f : 0.0f;
}

// ---------------------------------------------------------------------------
// RoPE + bf16 conversion of q,k (and plain bf16 of v).  One thread/element.
// ---------------------------------------------------------------------------
__global__ __launch_bounds__(256) void rope_kernel(
    const float* __restrict__ q, const float* __restrict__ k,
    const float* __restrict__ v, const long long* __restrict__ pos,
    __bf16* __restrict__ qb, __bf16* __restrict__ kb, __bf16* __restrict__ vb) {
  const int idx = blockIdx.x * 256 + threadIdx.x;   // t*HID + c
  const int t = idx >> 12;          // / 4096
  const int c = idx & (HID - 1);
  const int d = c & (HEADD - 1);
  const int j = d & 63;
  const float p = (float)pos[t];
  // inv_freq = 10000^(-j/64);  ln(10000) = 9.210340371976184
  const float ang = p * __expf(-(float)j * (9.210340371976184f / 64.0f));
  float sn, cs;
  __sincosf(ang, &sn, &cs);
  const int pidx = (d < 64) ? idx + 64 : idx - 64;
  const float sign = (d < 64) ? -1.0f : 1.0f;
  qb[idx] = (__bf16)(q[idx] * cs + sign * q[pidx] * sn);
  kb[idx] = (__bf16)(k[idx] * cs + sign * k[pidx] * sn);
  vb[idx] = (__bf16)v[idx];
}

// ---------------------------------------------------------------------------
// Generic WMMA GEMM:  C[M,N] = A[M,K](bf16) x W[N,K](fp32 -> bf16 in LDS)^T
// Tiles: 128x128x32, 8 waves (4 M-waves x 2 N-waves), wave tile 32x64.
// A staged via global_load_async_to_lds_b128; W converted through VGPRs.
// mode 0: outF = acc (+bias) (+resid)
// mode 3: outB = bf16( mask ? silu(gate)*acc : 0 )     (MLP "up" fusion)
// ---------------------------------------------------------------------------
#define GBM 128
#define GBN 128
#define GBK 32
#define GLD 40   // padded LDS row stride (bf16 elems): 80B rows, 16B aligned

__global__ __launch_bounds__(256) void gemm_kernel(
    const __bf16* __restrict__ A, const float* __restrict__ W,
    const float* __restrict__ bias, const float* __restrict__ resid,
    const float* __restrict__ gate, const unsigned char* __restrict__ mask,
    float* __restrict__ outF, __bf16* __restrict__ outB,
    int M, int N, int K, int mode) {
  __shared__ __bf16 As[GBM * GLD];
  __shared__ __bf16 Bs[GBN * GLD];
  const int tid = threadIdx.x;
  const int lane = tid & 31;
  const int wid = tid >> 5;
  const int waveM = wid >> 1;          // 0..3
  const int waveN = wid & 1;           // 0..1
  const int m0 = blockIdx.y * GBM;
  const int n0 = blockIdx.x * GBN;

  v8f acc[2][4];
#pragma unroll
  for (int i = 0; i < 2; ++i)
#pragma unroll
    for (int j = 0; j < 4; ++j)
#pragma unroll
      for (int e = 0; e < 8; ++e) acc[i][j][e] = 0.0f;

  // Cooperative load mapping: 2 threads per row, 16 elems each.
  const int lrow = tid >> 1;           // 0..127
  const int lcol = (tid & 1) * 16;     // 0 or 16
  const __bf16* ap = A + (size_t)(m0 + lrow) * K + lcol;
  const float* bp = W + (size_t)(n0 + lrow) * K + lcol;
  const unsigned a_lds = lds_off32(&As[lrow * GLD + lcol]);  // 32B per thread

  const int lm = lane & 15;
  const int khalf = (lane >> 4) * 8;   // A-frag K split per lane-half
  const int kb16 = (lane >> 4) * 16;   // unused; B-frag loads also use khalf

  for (int k0 = 0; k0 < K; k0 += GBK) {
    __syncthreads();
    // ---- stage A: async global->LDS (bf16 passthrough, no VGPR round-trip)
    async_ld_b128x2(a_lds, ap + k0);
    // ---- stage B (fp32 -> bf16 on the fly) ----
    float4 f0 = *(const float4*)(bp + k0);
    float4 f1 = *(const float4*)(bp + k0 + 4);
    float4 f2 = *(const float4*)(bp + k0 + 8);
    float4 f3 = *(const float4*)(bp + k0 + 12);
    v8bf b0, b1;
    b0[0] = (__bf16)f0.x; b0[1] = (__bf16)f0.y; b0[2] = (__bf16)f0.z; b0[3] = (__bf16)f0.w;
    b0[4] = (__bf16)f1.x; b0[5] = (__bf16)f1.y; b0[6] = (__bf16)f1.z; b0[7] = (__bf16)f1.w;
    b1[0] = (__bf16)f2.x; b1[1] = (__bf16)f2.y; b1[2] = (__bf16)f2.z; b1[3] = (__bf16)f2.w;
    b1[4] = (__bf16)f3.x; b1[5] = (__bf16)f3.y; b1[6] = (__bf16)f3.z; b1[7] = (__bf16)f3.w;
    *(v8bf*)&Bs[lrow * GLD + lcol] = b0;
    *(v8bf*)&Bs[lrow * GLD + lcol + 8] = b1;
    if (k0 + GBK < K) {                 // gfx1250 global_prefetch_b8
      __builtin_prefetch(ap + k0 + GBK, 0, 3);
      __builtin_prefetch(bp + k0 + GBK, 0, 3);
    }
    wait_async0();                      // s_wait_asynccnt 0
    __syncthreads();

    // ---- fragments + WMMA ----
    v16bf af[2];
#pragma unroll
    for (int mf = 0; mf < 2; ++mf) {
      const int row = waveM * 32 + mf * 16 + lm;
      v8bf lo = *(const v8bf*)&As[row * GLD + khalf];
      v8bf hi = *(const v8bf*)&As[row * GLD + khalf + 16];
      af[mf] = mkv16(lo, hi);
    }
#pragma unroll
    for (int nf = 0; nf < 4; ++nf) {
      const int col = waveN * 64 + nf * 16 + lm;
      v8bf lo = *(const v8bf*)&Bs[col * GLD + khalf];
      v8bf hi = *(const v8bf*)&Bs[col * GLD + khalf + 16];
      v16bf bfrag = mkv16(lo, hi);
      acc[0][nf] = wmma_bf16(af[0], bfrag, acc[0][nf]);
      acc[1][nf] = wmma_bf16(af[1], bfrag, acc[1][nf]);
    }
  }

  // ---- epilogue ----
  const int half8 = (lane >> 4) << 3;
#pragma unroll
  for (int mf = 0; mf < 2; ++mf)
#pragma unroll
    for (int nf = 0; nf < 4; ++nf)
#pragma unroll
      for (int r = 0; r < 8; ++r) {
        const int row = m0 + waveM * 32 + mf * 16 + r + half8;
        const int col = n0 + waveN * 64 + nf * 16 + lm;
        const size_t idx = (size_t)row * N + col;
        float v = acc[mf][nf][r];
        if (bias) v += bias[col];
        if (mode == 0) {
          if (resid) v += resid[idx];
          outF[idx] = v;
        } else {  // mode 3: inter = silu(gate) * up * topk_mask  -> bf16
          const float g = gate[idx];
          const float s = g / (1.0f + __expf(-g));
          outB[idx] = (__bf16)(mask[idx] ? s * v : 0.0f);
        }
      }
}

// ---------------------------------------------------------------------------
// Flash attention, one block = (head, 64-row q-tile), 4 waves x 16 q-rows.
// Q/K tiles staged with async global->LDS; V transposed through VGPRs.
// ---------------------------------------------------------------------------
#define FQLD 136   // 128 + 8 pad (272B rows)
#define FVLD 72    // 64 + 8 pad (144B rows)

__global__ __launch_bounds__(128) void flash_kernel(
    const __bf16* __restrict__ Qg, const __bf16* __restrict__ Kg,
    const __bf16* __restrict__ Vg, const float* __restrict__ head_mask,
    const int* __restrict__ amask, __bf16* __restrict__ Og) {
  __shared__ __bf16 Qs[64 * FQLD];
  __shared__ __bf16 Ks[64 * FQLD];
  __shared__ __bf16 Vts[128 * FVLD];        // transposed: [d][k]
  __shared__ __bf16 Ps[4 * 16 * FVLD];      // per-wave P scratch
  __shared__ int ams[64];

  const int h = blockIdx.x;
  const int qt = blockIdx.y;
  const int tid = threadIdx.x;
  const int lane = tid & 31;
  const int wid = tid >> 5;
  const int q0 = qt * 64;
  const int lr = tid >> 1;                  // 0..63
  const int lc = (tid & 1) * 64;            // 0 or 64
  const int lm = lane & 15;
  const int khalf = (lane >> 4) * 8;
  const int kb16 = (lane >> 4) * 16;
  const int half8 = (lane >> 4) << 3;

  // stage Q tile once (async; first in-loop wait covers it)
  const unsigned q_lds = lds_off32(&Qs[lr * FQLD + lc]);
  const unsigned k_lds = lds_off32(&Ks[lr * FQLD + lc]);
  async_ld_b128x8(q_lds, Qg + (size_t)(q0 + lr) * HID + h * HEADD + lc);

  v8f o[8], mrow, lrow;
#pragma unroll
  for (int i = 0; i < 8; ++i)
#pragma unroll
    for (int e = 0; e < 8; ++e) o[i][e] = 0.0f;
#pragma unroll
  for (int e = 0; e < 8; ++e) { mrow[e] = -1e30f; lrow[e] = 0.0f; }

  const int nkt = qt + 1;   // causal
  for (int kt = 0; kt < nkt; ++kt) {
    const int k0 = kt * 64;
    __syncthreads();
    {  // stage K (async, as-is) and V (transposed via VGPRs)
      async_ld_b128x8(k_lds, Kg + (size_t)(k0 + lr) * HID + h * HEADD + lc);
      const __bf16* vp = Vg + (size_t)(k0 + lr) * HID + h * HEADD + lc;
#pragma unroll
      for (int i = 0; i < 8; ++i) {
        v8bf vv = *(const v8bf*)(vp + i * 8);
#pragma unroll
        for (int jj = 0; jj < 8; ++jj)
          Vts[(lc + i * 8 + jj) * FVLD + lr] = vv[jj];
      }
      if (tid < 64) ams[tid] = amask[k0 + tid];
    }
    wait_async0();                      // Q (first iter) + K writes complete
    __syncthreads();

    // ---- S = Q K^T (16x64 per wave) ----
    v8f s[4];
#pragma unroll
    for (int nf = 0; nf < 4; ++nf)
#pragma unroll
      for (int e = 0; e < 8; ++e) s[nf][e] = 0.0f;
    const int mb = wid * 16;
#pragma unroll
    for (int ks = 0; ks < 4; ++ks) {
      const int row = mb + lm;
      v8bf alo = *(const v8bf*)&Qs[row * FQLD + ks * 32 + khalf];
      v8bf ahi = *(const v8bf*)&Qs[row * FQLD + ks * 32 + khalf + 16];
      v16bf a = mkv16(alo, ahi);
#pragma unroll
      for (int nf = 0; nf < 4; ++nf) {
        const int col = nf * 16 + lm;
        v8bf blo = *(const v8bf*)&Ks[col * FQLD + ks * 32 + khalf];
        v8bf bhi = *(const v8bf*)&Ks[col * FQLD + ks * 32 + khalf + 16];
        s[nf] = wmma_bf16(a, mkv16(blo, bhi), s[nf]);
      }
    }
    // ---- scale + causal/pad mask ----
    const float scale = 0.08838834764831845f;  // 1/sqrt(128)
#pragma unroll
    for (int nf = 0; nf < 4; ++nf)
#pragma unroll
      for (int r = 0; r < 8; ++r) {
        float v = s[nf][r] * scale;
        const int qi = q0 + mb + r + half8;
        const int krel = nf * 16 + lm;
        if (k0 + krel > qi || ams[krel] == 0) v = -1e30f;
        s[nf][r] = v;
      }
    // ---- online softmax (row stats live in C-fragment layout) ----
    v8f rmax;
#pragma unroll
    for (int r = 0; r < 8; ++r)
      rmax[r] = fmaxf(fmaxf(s[0][r], s[1][r]), fmaxf(s[2][r], s[3][r]));
#pragma unroll
    for (int x = 1; x <= 8; x <<= 1)
#pragma unroll
      for (int r = 0; r < 8; ++r)
        rmax[r] = fmaxf(rmax[r], __shfl_xor(rmax[r], x, 32));
    v8f alpha, rsum;
#pragma unroll
    for (int r = 0; r < 8; ++r) {
      const float mn = fmaxf(mrow[r], rmax[r]);
      alpha[r] = __expf(mrow[r] - mn);
      mrow[r] = mn;
      rsum[r] = 0.0f;
    }
#pragma unroll
    for (int nf = 0; nf < 4; ++nf)
#pragma unroll
      for (int r = 0; r < 8; ++r) {
        const float p = __expf(s[nf][r] - mrow[r]);
        s[nf][r] = p;
        rsum[r] += p;
      }
#pragma unroll
    for (int x = 1; x <= 8; x <<= 1)
#pragma unroll
      for (int r = 0; r < 8; ++r) rsum[r] += __shfl_xor(rsum[r], x, 32);
#pragma unroll
    for (int r = 0; r < 8; ++r) lrow[r] = lrow[r] * alpha[r] + rsum[r];
#pragma unroll
    for (int i = 0; i < 8; ++i)
#pragma unroll
      for (int r = 0; r < 8; ++r) o[i][r] *= alpha[r];

    // ---- P (C-layout) -> LDS -> A-fragments; O += P x Vt ----
    __bf16* Pw = &Ps[wid * 16 * FVLD];
#pragma unroll
    for (int nf = 0; nf < 4; ++nf)
#pragma unroll
      for (int r = 0; r < 8; ++r)
        Pw[(r + half8) * FVLD + nf * 16 + lm] = (__bf16)s[nf][r];
    // same-wave DS ops are in order; no cross-wave sharing of Pw
#pragma unroll
    for (int ks = 0; ks < 2; ++ks) {
      v8bf plo = *(const v8bf*)&Pw[lm * FVLD + ks * 32 + khalf];
      v8bf phi = *(const v8bf*)&Pw[lm * FVLD + ks * 32 + khalf + 16];
      v16bf a = mkv16(plo, phi);
#pragma unroll
      for (int nf = 0; nf < 8; ++nf) {
        const int d = nf * 16 + lm;
        v8bf blo = *(const v8bf*)&Vts[d * FVLD + ks * 32 + khalf];
        v8bf bhi = *(const v8bf*)&Vts[d * FVLD + ks * 32 + khalf + 16];
        o[nf] = wmma_bf16(a, mkv16(blo, bhi), o[nf]);
      }
    }
  }

  // ---- epilogue: normalize, head mask, write bf16 ----
#pragma unroll
  for (int r = 0; r < 8; ++r) {
    const int t = q0 + wid * 16 + r + half8;
    const float inv = (lrow[r] > 0.0f) ? (1.0f / lrow[r]) : 0.0f;
    const float sc = inv * head_mask[t * NHEAD + h];
#pragma unroll
    for (int nf = 0; nf < 8; ++nf)
      Og[(size_t)t * HID + h * HEADD + nf * 16 + lm] = (__bf16)(o[nf][r] * sc);
  }
}

// ---------------------------------------------------------------------------
// MLP top-k (2048 of 11008) per row: stage bit-ordered keys in LDS, 32-step
// MSB-first bisection for the k-th largest, then write byte mask.
// ---------------------------------------------------------------------------
__global__ __launch_bounds__(256) void mlp_select_kernel(
    const float* __restrict__ logits, unsigned char* __restrict__ mask) {
  __shared__ unsigned int keys[ISZ];   // 43KB
  __shared__ int cnt;
  __shared__ int taken;
  const int t = blockIdx.x;
  const float* lr = logits + (size_t)t * ISZ;
  for (int i = threadIdx.x; i < ISZ; i += 256) {
    const unsigned int u = __float_as_uint(lr[i]);
    keys[i] = (u & 0x80000000u) ? ~u : (u | 0x80000000u);  // order-preserving
  }
  __syncthreads();
  unsigned int prefix = 0u;
  for (int b = 31; b >= 0; --b) {
    const unsigned int cand = prefix | (1u << b);
    if (threadIdx.x == 0) cnt = 0;
    __syncthreads();
    int local = 0;
    for (int i = threadIdx.x; i < ISZ; i += 256) local += (keys[i] >= cand) ? 1 : 0;
    atomicAdd(&cnt, local);
    __syncthreads();
    if (cnt >= MLPK) prefix = cand;
    __syncthreads();
  }
  if (threadIdx.x == 0) { cnt = 0; taken = 0; }
  __syncthreads();
  int lg = 0;
  for (int i = threadIdx.x; i < ISZ; i += 256) lg += (keys[i] > prefix) ? 1 : 0;
  atomicAdd(&cnt, lg);
  __syncthreads();
  const int need = MLPK - cnt;     // ties at threshold to admit (rare)
  unsigned char* mr = mask + (size_t)t * ISZ;
  for (int i = threadIdx.x; i < ISZ; i += 256) {
    const unsigned int ky = keys[i];
    unsigned char m = 0;
    if (ky > prefix) m = 1;
    else if (ky == prefix) { if (atomicAdd(&taken, 1) < need) m = 1; }
    mr[i] = m;
  }
}

// ---------------------------------------------------------------------------
// Host launcher
// ---------------------------------------------------------------------------
extern "C" void kernel_launch(void* const* d_in, const int* in_sizes, int n_in,
                              void* d_out, int out_size, void* d_ws, size_t ws_size,
                              hipStream_t stream) {
  (void)in_sizes; (void)n_in; (void)out_size; (void)ws_size;
  const float*      hidden = (const float*)d_in[0];
  const int*        amask  = (const int*)d_in[1];
  const long long*  pos    = (const long long*)d_in[2];
  const float*      wq     = (const float*)d_in[3];
  const float*      wk     = (const float*)d_in[4];
  const float*      wv     = (const float*)d_in[5];
  const float*      wo     = (const float*)d_in[6];
  const float*      w_gate = (const float*)d_in[7];
  const float*      w_up   = (const float*)d_in[8];
  const float*      w_down = (const float*)d_in[9];
  const float*      ln1    = (const float*)d_in[10];
  const float*      ln2    = (const float*)d_in[11];
  const float*      apw    = (const float*)d_in[12];
  const float*      apb    = (const float*)d_in[13];
  const float*      mpw    = (const float*)d_in[14];
  const float*      mpb    = (const float*)d_in[15];
  float* out = (float*)d_out;

  char* ws = (char*)d_ws;
  size_t off = 0;
  auto alloc = [&](size_t b) { char* p = ws + off; off += (b + 255) & ~(size_t)255; return p; };

  __bf16* hs1n   = (__bf16*)alloc((size_t)T_SEQ * HID * 2);
  float*  qf     = (float*) alloc((size_t)T_SEQ * HID * 4);   // q,k,v contiguous
  float*  kf     = (float*) alloc((size_t)T_SEQ * HID * 4);
  float*  vf     = (float*) alloc((size_t)T_SEQ * HID * 4);
  __bf16* qb     = (__bf16*)alloc((size_t)T_SEQ * HID * 2);   // qb,kb,vb contiguous
  __bf16* kb     = (__bf16*)alloc((size_t)T_SEQ * HID * 2);
  __bf16* vb     = (__bf16*)alloc((size_t)T_SEQ * HID * 2);
  float*  hmask  = (float*) alloc((size_t)T_SEQ * NHEAD * 4);
  __bf16* attnb  = (__bf16*)alloc((size_t)T_SEQ * HID * 2);
  float*  hs2    = (float*) alloc((size_t)T_SEQ * HID * 4);
  __bf16* hs2n   = (__bf16*)alloc((size_t)T_SEQ * HID * 2);
  float*  gatebuf= (float*) alloc((size_t)T_SEQ * ISZ * 4);
  // Reuse (regions dead by the time these are live):
  float*         mlogit = qf;                      // 90.2MB <= 100.7MB q/k/v
  unsigned char* fmask  = (unsigned char*)qb;      // 22.5MB <= 50.3MB qb/kb/vb
  __bf16*        interb = (__bf16*)qf;             // 45.1MB, after mlogit used

  const dim3 gH(HID / 128, T_SEQ / 128);
  const dim3 gI(ISZ / 128, T_SEQ / 128);

  // 1) hs = rms_norm(x, ln1) -> bf16
  rmsnorm_kernel<<<T_SEQ, 256, 0, stream>>>(hidden, ln1, hs1n, HID);
  // 2) head top-k mask
  attn_pred_kernel<<<T_SEQ, 32, 0, stream>>>(hs1n, apw, apb, hmask);
  // 3) q,k,v projections (WMMA)
  gemm_kernel<<<gH, 256, 0, stream>>>(hs1n, wq, nullptr, nullptr, nullptr, nullptr,
                                      qf, nullptr, T_SEQ, HID, HID, 0);
  gemm_kernel<<<gH, 256, 0, stream>>>(hs1n, wk, nullptr, nullptr, nullptr, nullptr,
                                      kf, nullptr, T_SEQ, HID, HID, 0);
  gemm_kernel<<<gH, 256, 0, stream>>>(hs1n, wv, nullptr, nullptr, nullptr, nullptr,
                                      vf, nullptr, T_SEQ, HID, HID, 0);
  // 4) RoPE + bf16 convert
  rope_kernel<<<(T_SEQ * HID) / 256, 256, 0, stream>>>(qf, kf, vf, pos, qb, kb, vb);
  // 5) flash attention + head mask
  flash_kernel<<<dim3(NHEAD, T_SEQ / 64), 128, 0, stream>>>(qb, kb, vb, hmask, amask, attnb);
  // 6) wo projection + residual
  gemm_kernel<<<gH, 256, 0, stream>>>(attnb, wo, nullptr, hidden, nullptr, nullptr,
                                      hs2, nullptr, T_SEQ, HID, HID, 0);
  // 7) second rmsnorm
  rmsnorm_kernel<<<T_SEQ, 256, 0, stream>>>(hs2, ln2, hs2n, HID);
  // 8) MLP predictor logits (+bias)
  gemm_kernel<<<gI, 256, 0, stream>>>(hs2n, mpw, mpb, nullptr, nullptr, nullptr,
                                      mlogit, nullptr, T_SEQ, ISZ, HID, 0);
  // 9) top-2048 mask per row
  mlp_select_kernel<<<T_SEQ, 256, 0, stream>>>(mlogit, fmask);
  // 10) gate projection
  gemm_kernel<<<gI, 256, 0, stream>>>(hs2n, w_gate, nullptr, nullptr, nullptr, nullptr,
                                      gatebuf, nullptr, T_SEQ, ISZ, HID, 0);
  // 11) up projection fused with silu(gate)*mask -> bf16 inter
  gemm_kernel<<<gI, 256, 0, stream>>>(hs2n, w_up, nullptr, nullptr, gatebuf, fmask,
                                      nullptr, interb, T_SEQ, ISZ, HID, 3);
  // 12) down projection + residual -> out
  gemm_kernel<<<gH, 256, 0, stream>>>(interb, w_down, nullptr, hs2, nullptr, nullptr,
                                      out, nullptr, T_SEQ, HID, ISZ, 0);
}
